// EdgeClassifier_61151744360719
// MI455X (gfx1250) — matrix operand order repaired and checked by reading, not compiled, not run on hardware
//
#include <hip/hip_runtime.h>

// ---------------------------------------------------------------------------
// Fully fused edge-MLP classifier for MI455X (gfx1250, wave32, WMMA).
// Roofline: ~295 GFLOP vs ~60 MB HBM traffic -> compute bound.
// Strategy: all weights f16 fragment-packed in ~300KB LDS (CDNA5 320KB/WGP),
// per-wave 16-edge tiles, whole MLP chain on v_wmma_f32_16x16x32_f16.
// v2: 16 waves / 512 threads per workgroup (4 waves per SIMD32) to hide
// LDS->WMMA chain latency; LDS total 307.75KB still fits one WGP.
// ---------------------------------------------------------------------------

#define HC        128
#define WAVES     16
#define NTHREADS  512

typedef __attribute__((ext_vector_type(16))) _Float16 v16h;
typedef __attribute__((ext_vector_type(8)))  float    v8f;
typedef __attribute__((ext_vector_type(4)))  float    v4f;

// ---- LDS layout (byte offsets into dynamic shared memory) -----------------
// Weight matrices packed "fragment-major": for a KxN weight, tile (kt,nt) is a
// 32x16 B-operand stored as [lane 0..31][16 f16] (32B per lane), so a WMMA B
// fragment is a single contiguous 32B LDS read per lane.
#define OFF_D1     0u        // [128,256] -> 4*16 tiles * 1KB = 64KB
#define OFF_P1     65536u    // 64KB
#define OFF_D2     131072u   // [256,128] -> 8*8 tiles        = 64KB
#define OFF_P2     196608u   // 64KB
#define OFF_L1     262144u   // [256, 64] -> 8*4 tiles        = 32KB
#define OFF_B1D    294912u   // 256 f32
#define OFF_B1P    295936u   // 256 f32
#define OFF_B2D    296960u   // 128 f32
#define OFF_B2P    297472u   // 128 f32
#define OFF_L1B    297984u   // 64 f32
#define OFF_L2W    298240u   // 64 f32
#define OFF_L2B    298496u   // 1 f32 (padded)
#define OFF_CHUNK  298752u   // per-wave 16x32 f16 reshape chunk: 16 * 1KB
#define SMEM_TOTAL (OFF_CHUNK + WAVES * 1024u)   // 315136B = 307.75KB < 320KB

__device__ __forceinline__ v8f wmma_f16(v16h a, v16h b, v8f c) {
  // D = A(16x32 f16) * B(32x16 f16) + C(16x16 f32)
  return __builtin_amdgcn_wmma_f32_16x16x32_f16(
      /*neg_a=*/false, a, /*neg_b=*/false, b,
      /*c_mod=*/(short)0, c, /*reuse_a=*/false, /*reuse_b=*/false);
}

// Convert a row-major KxN f32 weight into fragment-packed f16 tiles in LDS.
// Element (k,n) -> tile (k>>5)*NT + (n>>4); within tile:
//   lane = (n&15) | (((k>>3)&1)<<4), slot = (k&7) | (((k>>4)&1)<<3)  (matches 16-bit B layout).
__device__ __forceinline__ void pack_weight(_Float16* lds16, const float* W,
                                            int K, int N, int logN,
                                            unsigned offH, int tid) {
  const int NT = N >> 4;
  const int total = K << logN;
  for (int i = tid; i < total; i += NTHREADS) {
    const int k = i >> logN;
    const int n = i & (N - 1);
    const int tileIdx = (k >> 5) * NT + (n >> 4);
    const int ln      = (n & 15) | (((k >> 3) & 1) << 4);
    const int slot    = (k & 7) | (((k >> 4) & 1) << 3);
    lds16[offH + (unsigned)((tileIdx << 9) + (ln << 4) + slot)] = (_Float16)W[i];
  }
}

extern "C" __global__ __launch_bounds__(NTHREADS, 1)
void edgecls_fused_wmma(const float* __restrict__ z_drug,
                        const float* __restrict__ z_protein,
                        const int*   __restrict__ eli,
                        const float* __restrict__ d1W, const float* __restrict__ d1b,
                        const float* __restrict__ d2W, const float* __restrict__ d2b,
                        const float* __restrict__ p1W, const float* __restrict__ p1b,
                        const float* __restrict__ p2W, const float* __restrict__ p2b,
                        const float* __restrict__ l1W, const float* __restrict__ l1b,
                        const float* __restrict__ l2W, const float* __restrict__ l2b,
                        float* __restrict__ out,
                        int E, int nTiles) {
  extern __shared__ char smem[];
  _Float16* lds16 = (_Float16*)smem;
  float*    ldsF  = (float*)smem;

  const int tid = threadIdx.x;

  // ---- one-time cooperative weight preload (f32 -> f16 fragment packing) --
  pack_weight(lds16, d1W, 128, 256, 8, OFF_D1 >> 1, tid);
  pack_weight(lds16, p1W, 128, 256, 8, OFF_P1 >> 1, tid);
  pack_weight(lds16, d2W, 256, 128, 7, OFF_D2 >> 1, tid);
  pack_weight(lds16, p2W, 256, 128, 7, OFF_P2 >> 1, tid);
  pack_weight(lds16, l1W, 256,  64, 6, OFF_L1 >> 1, tid);
  for (int i = tid; i < 256; i += NTHREADS) ldsF[(OFF_B1D >> 2) + i] = d1b[i];
  for (int i = tid; i < 256; i += NTHREADS) ldsF[(OFF_B1P >> 2) + i] = p1b[i];
  for (int i = tid; i < 128; i += NTHREADS) ldsF[(OFF_B2D >> 2) + i] = d2b[i];
  for (int i = tid; i < 128; i += NTHREADS) ldsF[(OFF_B2P >> 2) + i] = p2b[i];
  for (int i = tid; i <  64; i += NTHREADS) ldsF[(OFF_L1B >> 2) + i] = l1b[i];
  for (int i = tid; i <  64; i += NTHREADS) ldsF[(OFF_L2W >> 2) + i] = l2W[i];
  if (tid == 0) ldsF[OFF_L2B >> 2] = l2b[0];
  __syncthreads();

  const int lane = tid & 31;
  const int wIn  = tid >> 5;
  const int col  = lane & 15;   // WMMA C/D column for this lane
  const int hi   = lane >> 4;   // 0: rows 0..7, 1: rows 8..15 (C layout)
  _Float16* chunk = lds16 + (OFF_CHUNK >> 1) + wIn * 512;   // 16x32 f16
  const float* l1bL = ldsF + (OFF_L1B >> 2);
  const float* l2wL = ldsF + (OFF_L2W >> 2);
  const float  l2bias = ldsF[OFF_L2B >> 2];

  const long numWaves = (long)gridDim.x * WAVES;

  for (long tile = (long)blockIdx.x * WAVES + wIn; tile < nTiles;
       tile += numWaves) {
    const int eBase = (int)tile * 16;

    // head accumulators: 16x64 f32 across both concat halves
    v8f hAcc[4];
#pragma unroll
    for (int j = 0; j < 4; ++j) {
#pragma unroll
      for (int v = 0; v < 8; ++v) hAcc[j][v] = 0.f;
    }

#pragma unroll 1
    for (int br = 0; br < 2; ++br) {   // 0 = drug branch, 1 = protein branch
      const float* zb = br ? z_protein : z_drug;
      const int*   ip = eli + (br ? E : 0);
      const unsigned w1o = (br ? OFF_P1 : OFF_D1) >> 1;
      const unsigned w2o = (br ? OFF_P2 : OFF_D2) >> 1;
      const float* b1 = ldsF + ((br ? OFF_B1P : OFF_B1D) >> 2);
      const float* b2 = ldsF + ((br ? OFF_B2P : OFF_B2D) >> 2);

      // ---- gather: build A fragments (16 edges x 128 feat) directly ------
      // 16-bit A layout: lane<16 holds K {0..7,16..23}+32kt, lane>=16 holds
      // K {8..15,24..31}+32kt  -> two 32B global reads per K-tile per lane.
      int e = eBase + col;
      if (e >= E) e = E - 1;
      const int r = ip[e];
      const float* src = zb + (size_t)r * HC;
      const int koff = hi << 3;

      v16h A1[4];
#pragma unroll
      for (int kt = 0; kt < 4; ++kt) {
        const float* s0 = src + kt * 32 + koff;
        v4f a0 = *(const v4f*)(s0);
        v4f a1 = *(const v4f*)(s0 + 4);
        v4f c0 = *(const v4f*)(s0 + 16);
        v4f c1 = *(const v4f*)(s0 + 20);
        v16h A;
#pragma unroll
        for (int q = 0; q < 4; ++q) {
          A[q]      = (_Float16)a0[q];
          A[4 + q]  = (_Float16)a1[q];
          A[8 + q]  = (_Float16)c0[q];
          A[12 + q] = (_Float16)c1[q];
        }
        A1[kt] = A;
      }

      // h2 accumulators (16x128), seeded with layer-2 bias
      v8f acc2[8];
#pragma unroll
      for (int j = 0; j < 8; ++j) {
        const float bv = b2[j * 16 + col];
#pragma unroll
        for (int v = 0; v < 8; ++v) acc2[j][v] = bv;
      }

      // ---- GEMM1 (128->256) fused into GEMM2 (256->128) ------------------
      // Produce h1 two n-tiles at a time (one 16x32 A-slice), reshape via
      // the per-wave LDS chunk, and immediately fold into acc2.
#pragma unroll 1
      for (int p = 0; p < 8; ++p) {
#pragma unroll
        for (int t = 0; t < 2; ++t) {
          const int nt = p * 2 + t;
          const float bv = b1[nt * 16 + col];
          v8f c1v;
#pragma unroll
          for (int v = 0; v < 8; ++v) c1v[v] = bv;
#pragma unroll
          for (int kt = 0; kt < 4; ++kt) {
            v16h B = *(const v16h*)(lds16 + w1o +
                                    (unsigned)(((kt * 16 + nt) << 9) + (lane << 4)));
            c1v = wmma_f16(A1[kt], B, c1v);
          }
          // relu + f16 + scatter C-layout -> A-fragment layout in chunk
          const int kk   = (t << 4) + col;
          const int slot = (kk & 7) | ((kk >> 4) << 3);
          const int tl   = ((kk >> 3) & 1) << 4;
#pragma unroll
          for (int v = 0; v < 8; ++v) {
            float x = c1v[v];
            x = x > 0.f ? x : 0.f;
            const int m = v + (hi << 3);
            chunk[(tl + m) * 16 + slot] = (_Float16)x;
          }
        }
        __builtin_amdgcn_wave_barrier();   // same-wave LDS ops are in-order
        v16h Ac = *(const v16h*)(chunk + (lane << 4));
#pragma unroll
        for (int j = 0; j < 8; ++j) {
          v16h B = *(const v16h*)(lds16 + w2o +
                                  (unsigned)(((p * 8 + j) << 9) + (lane << 4)));
          acc2[j] = wmma_f16(Ac, B, acc2[j]);
        }
        __builtin_amdgcn_wave_barrier();
      }

      // ---- head GEMM: relu(h2) slice -> l1 (this branch's concat half) ---
#pragma unroll 1
      for (int q = 0; q < 4; ++q) {
#pragma unroll
        for (int t = 0; t < 2; ++t) {
          const int j    = q * 2 + t;
          const int kk   = (t << 4) + col;
          const int slot = (kk & 7) | ((kk >> 4) << 3);
          const int tl   = ((kk >> 3) & 1) << 4;
#pragma unroll
          for (int v = 0; v < 8; ++v) {
            float x = acc2[j][v];
            x = x > 0.f ? x : 0.f;
            const int m = v + (hi << 3);
            chunk[(tl + m) * 16 + slot] = (_Float16)x;
          }
        }
        __builtin_amdgcn_wave_barrier();
        v16h Ac = *(const v16h*)(chunk + (lane << 4));
        const int kt = q + (br << 2);   // drug: K 0..127, protein: K 128..255
#pragma unroll
        for (int hn = 0; hn < 4; ++hn) {
          v16h B = *(const v16h*)(lds16 + (OFF_L1 >> 1) +
                                  (unsigned)(((kt * 4 + hn) << 9) + (lane << 4)));
          hAcc[hn] = wmma_f16(Ac, B, hAcc[hn]);
        }
        __builtin_amdgcn_wave_barrier();
      }
    } // branches

    // ---- final: out = relu(hAcc + l1_b) . l2_W + l2_b --------------------
    // C layout: lane holds col = lane&15, rows v + 8*(lane>>4). Reduce the
    // 64-wide dot across the 16 lanes of each half-wave with shfl_xor.
#pragma unroll
    for (int v = 0; v < 8; ++v) {
      float s = 0.f;
#pragma unroll
      for (int hn = 0; hn < 4; ++hn) {
        const int n = hn * 16 + col;
        float x = hAcc[hn][v] + l1bL[n];
        x = x > 0.f ? x : 0.f;
        s += x * l2wL[n];
      }
      s += __shfl_xor(s, 1);
      s += __shfl_xor(s, 2);
      s += __shfl_xor(s, 4);
      s += __shfl_xor(s, 8);
      if (col == 0) {
        const int e = eBase + v + (hi << 3);
        if (e < E) out[e] = s + l2bias;
      }
    }
  }
}

extern "C" void kernel_launch(void* const* d_in, const int* in_sizes, int n_in,
                              void* d_out, int out_size, void* d_ws, size_t ws_size,
                              hipStream_t stream) {
  (void)n_in; (void)d_ws; (void)ws_size; (void)out_size;
  const float* z_drug    = (const float*)d_in[0];
  const float* z_protein = (const float*)d_in[1];
  const int*   eli       = (const int*)d_in[2];
  const float* d1W = (const float*)d_in[3];
  const float* d1b = (const float*)d_in[4];
  const float* d2W = (const float*)d_in[5];
  const float* d2b = (const float*)d_in[6];
  const float* p1W = (const float*)d_in[7];
  const float* p1b = (const float*)d_in[8];
  const float* p2W = (const float*)d_in[9];
  const float* p2b = (const float*)d_in[10];
  const float* l1W = (const float*)d_in[11];
  const float* l1b = (const float*)d_in[12];
  const float* l2W = (const float*)d_in[13];
  const float* l2b = (const float*)d_in[14];
  float* out = (float*)d_out;

  const int E      = in_sizes[2] / 2;        // edge_label_index is [2, E]
  const int nTiles = (E + 15) / 16;

  int blocks = 256;                           // persistent-ish: amortize preload
  const int maxB = (nTiles + WAVES - 1) / WAVES;
  if (blocks > maxB) blocks = maxB;
  if (blocks < 1) blocks = 1;

  edgecls_fused_wmma<<<dim3(blocks), dim3(NTHREADS), SMEM_TOTAL, stream>>>(
      z_drug, z_protein, eli,
      d1W, d1b, d2W, d2b, p1W, p1b, p2W, p2b, l1W, l1b, l2W, l2b,
      out, E, nTiles);
}
